// GATModel_7636451852532
// MI455X (gfx1250) — matrix operand order repaired and checked by reading, not compiled, not run on hardware
//
#include <hip/hip_runtime.h>
#include <hip/hip_bf16.h>
#include <math.h>

// ---------------- problem constants ----------------
static constexpr int kN   = 4096;
static constexpr int kDF  = 158;
static constexpr int kDFP = 160;   // K padded to multiple of 32
static constexpr int kT   = 60;
static constexpr int kH   = 64;
static constexpr int kG   = 192;   // 3*H
static constexpr int kRows = 32;   // rows per block in GRU kernel

typedef __attribute__((ext_vector_type(16))) _Float16 v16h;
typedef __attribute__((ext_vector_type(8)))  float    v8f;

__device__ __forceinline__ v8f wmma16(v16h a, v16h b, v8f c) {
  return __builtin_amdgcn_wmma_f32_16x16x32_f16(false, a, false, b, (short)0, c,
                                                false, false);
}

__device__ __forceinline__ float sigm_(float x) { return 1.0f / (1.0f + __expf(-x)); }
__device__ __forceinline__ float leaky_(float x) { return x >= 0.0f ? x : 0.01f * x; }

// ---------------- gfx1250 async global->LDS helpers (inline asm, ASYNCcnt) ----------------
__device__ __forceinline__ unsigned lds_off_u32(const void* p) {
  // generic pointers to LDS: addr[31:0] is the wave-relative LDS byte offset
  return (unsigned)(uintptr_t)p;
}
__device__ __forceinline__ void async_g2l_b32(unsigned lds_addr, unsigned voff_bytes,
                                              const void* sbase) {
  // GVS mode: mem = SADDR(64b sgpr) + VADDR(32b vgpr); LDS dest = VGPR[vdst]
  asm volatile("global_load_async_to_lds_b32 %0, %1, %2"
               :: "v"(lds_addr), "v"(voff_bytes), "s"(sbase)
               : "memory");
}
__device__ __forceinline__ void wait_async0() {
  asm volatile("s_wait_asynccnt 0x0" ::: "memory");
}

// ---------------- WMMA fragment loaders ----------------
// A (16x32 f16): lane<16 -> row=lane, K={0..15}; lane>=16 -> row=lane-16, K={16..31}
__device__ __forceinline__ v16h load_a_lds(const _Float16* base, int ld) {
  const int lane = threadIdx.x & 31;
  const _Float16* p = base + (lane & 15) * ld + ((lane >> 4) << 4);
  v16h a;
#pragma unroll
  for (int i = 0; i < 8; ++i) { a[i] = p[i]; a[i + 8] = p[8 + i]; }
  return a;
}
// A fragment built from f32 LDS (async-staged tile), converting during load
__device__ __forceinline__ v16h load_a_lds_f32(const float* base, int ld) {
  const int lane = threadIdx.x & 31;
  const float* p = base + (lane & 15) * ld + ((lane >> 4) << 4);
  v16h a;
#pragma unroll
  for (int i = 0; i < 8; ++i) { a[i] = (_Float16)p[i]; a[i + 8] = (_Float16)p[8 + i]; }
  return a;
}
// B (32x16) from N-major storage (W[n][k] => B[k][n]); f16 LDS
__device__ __forceinline__ v16h load_b_nmajor(const _Float16* base, int ld) {
  const int lane = threadIdx.x & 31;
  const _Float16* p = base + (lane & 15) * ld + ((lane >> 4) << 4);
  v16h b;
#pragma unroll
  for (int i = 0; i < 16; ++i) b[i] = p[i];
  return b;
}
// B (32x16) from N-major f32 global weights (contiguous 64B per lane, L2-resident)
__device__ __forceinline__ v16h load_b_nmajor_g(const float* __restrict__ base, int ld) {
  const int lane = threadIdx.x & 31;
  const float* p = base + (lane & 15) * ld + ((lane >> 4) << 4);
  v16h b;
#pragma unroll
  for (int i = 0; i < 16; ++i) b[i] = (_Float16)p[i];
  return b;
}
// B (32x16) from K-major f32 LDS (async-staged hidden chunk)
__device__ __forceinline__ v16h load_b_kmajor_f32(const float* base, int ld) {
  const int lane = threadIdx.x & 31;
  const int col  = lane & 15;
  const int kb   = (lane >> 4) << 4;
  v16h b;
#pragma unroll
  for (int i = 0; i < 16; ++i) b[i] = (_Float16)base[(kb + i) * ld + col];
  return b;
}

// ======================= Kernel 1: fused 2-layer GRU (32 rows/block) ===================
__global__ __launch_bounds__(256) void gru_fused_kernel(
    const float* __restrict__ x,
    const float* __restrict__ Wih0, const float* __restrict__ Whh0,
    const float* __restrict__ bih0, const float* __restrict__ bhh0,
    const float* __restrict__ Wih1, const float* __restrict__ Whh1,
    const float* __restrict__ bih1, const float* __restrict__ bhh1,
    const float* __restrict__ Wt,   const float* __restrict__ bt,
    const float* __restrict__ avec,
    float* __restrict__ hidden, float* __restrict__ s_src, float* __restrict__ s_dst)
{
  __shared__ __attribute__((aligned(16))) _Float16 sWih0[kG * kDFP]; // 60 KB (aliased f32 hp later)
  __shared__ _Float16 sWhh0[kG * kH];                                // 24 KB
  __shared__ _Float16 sWih1[kG * kH];                                // 24 KB
  __shared__ __attribute__((aligned(16))) float sXf[kRows * kDFP];   // 20 KB async x_t tile (f32)
  __shared__ _Float16 sH0[kRows * kH];                               // 4 KB
  __shared__ _Float16 sH1[kRows * kH];                               // 4 KB

  const int tid  = threadIdx.x;
  const int wave = tid >> 5;
  const int lane = tid & 31;
  const int n0   = blockIdx.x * kRows;

  const int mt      = wave >> 2;           // 0..1 row tile
  const int nt      = wave & 3;            // 0..3 col tile (within a gate)
  const int colL    = lane & 15;
  const int rowBase = mt * 16 + ((lane >> 4) << 3);

  const unsigned xlds = lds_off_u32(sXf);

  // ---- kick off async stage of x_0 (overlaps with weight staging below) ----
  {
    const float* sb = x + (size_t)n0 * (kDF * kT) + 0;
    for (int idx = tid; idx < kRows * kDF; idx += 256) {
      int r = idx / kDF, f = idx - r * kDF;
      async_g2l_b32(xlds + (unsigned)(r * kDFP + f) * 4u,
                    (unsigned)(r * (kDF * kT) + f * kT) * 4u, sb);
    }
  }

  // ---- stage weights (f16) ----
  for (int idx = tid; idx < kG * kDFP; idx += 256) {
    int r = idx / kDFP, c = idx - r * kDFP;
    sWih0[idx] = (_Float16)(c < kDF ? Wih0[r * kDF + c] : 0.0f);
  }
  for (int idx = tid; idx < kG * kH; idx += 256) {
    sWhh0[idx] = (_Float16)Whh0[idx];
    sWih1[idx] = (_Float16)Wih1[idx];
  }
  for (int idx = tid; idx < kRows * kH; idx += 256) {
    sH0[idx] = (_Float16)0.0f;
    sH1[idx] = (_Float16)0.0f;
  }
  // zero the K-pad columns of the x tile (async never writes them)
  if (tid < kRows * 2) {
    int r = tid >> 1;
    sXf[r * kDFP + kDF + (tid & 1)] = 0.0f;
  }

  // ---- hoist per-lane gate biases ----
  const int cB = nt * 16 + colL;
  const float bR0  = bih0[cB] + bhh0[cB];
  const float bZ0  = bih0[kH + cB] + bhh0[kH + cB];
  const float bNi0 = bih0[2 * kH + cB];
  const float bNh0 = bhh0[2 * kH + cB];
  const float bR1  = bih1[cB] + bhh1[cB];
  const float bZ1  = bih1[kH + cB] + bhh1[kH + cB];
  const float bNi1 = bih1[2 * kH + cB];
  const float bNh1 = bhh1[2 * kH + cB];

  for (int t = 0; t < kT; ++t) {
    wait_async0();
    __syncthreads();                       // x_t landed everywhere; LDS stores visible

    // consume x_t + h0 into register fragments, then free the x buffer
    v16h aX[5], aH[2];
#pragma unroll
    for (int k = 0; k < 5; ++k) aX[k] = load_a_lds_f32(sXf + mt * 16 * kDFP + k * 32, kDFP);
#pragma unroll
    for (int k = 0; k < 2; ++k) aH[k] = load_a_lds(sH0 + mt * 16 * kH + k * 32, kH);
    __syncthreads();                       // all waves consumed sXf

    // overlap: async stage x_{t+1} while both GRU layers compute
    if (t + 1 < kT) {
      const float* sb = x + (size_t)n0 * (kDF * kT) + (t + 1);
      for (int idx = tid; idx < kRows * kDF; idx += 256) {
        int r = idx / kDF, f = idx - r * kDF;
        async_g2l_b32(xlds + (unsigned)(r * kDFP + f) * 4u,
                      (unsigned)(r * (kDF * kT) + f * kT) * 4u, sb);
      }
    }

    // ================= layer 0 =================
    float hn0[8];
    {
      v8f ar, az, ain, ahn;
#pragma unroll
      for (int v = 0; v < 8; ++v) { ar[v] = bR0; az[v] = bZ0; ain[v] = bNi0; ahn[v] = bNh0; }
#pragma unroll
      for (int k = 0; k < 5; ++k) {
        ar  = wmma16(aX[k], load_b_nmajor(sWih0 + (nt * 16) * kDFP + k * 32, kDFP), ar);
        az  = wmma16(aX[k], load_b_nmajor(sWih0 + (kH + nt * 16) * kDFP + k * 32, kDFP), az);
        ain = wmma16(aX[k], load_b_nmajor(sWih0 + (2 * kH + nt * 16) * kDFP + k * 32, kDFP), ain);
      }
#pragma unroll
      for (int k = 0; k < 2; ++k) {
        ar  = wmma16(aH[k], load_b_nmajor(sWhh0 + (nt * 16) * kH + k * 32, kH), ar);
        az  = wmma16(aH[k], load_b_nmajor(sWhh0 + (kH + nt * 16) * kH + k * 32, kH), az);
        ahn = wmma16(aH[k], load_b_nmajor(sWhh0 + (2 * kH + nt * 16) * kH + k * 32, kH), ahn);
      }
#pragma unroll
      for (int v = 0; v < 8; ++v) {
        float r_ = sigm_(ar[v]);
        float z_ = sigm_(az[v]);
        float n_ = tanhf(ain[v] + r_ * ahn[v]);
        float hp = (float)sH0[(rowBase + v) * kH + nt * 16 + colL];
        hn0[v] = (1.0f - z_) * n_ + z_ * hp;
      }
    }
    __syncthreads();
#pragma unroll
    for (int v = 0; v < 8; ++v)
      sH0[(rowBase + v) * kH + nt * 16 + colL] = (_Float16)hn0[v];
    __syncthreads();

    // ================= layer 1 =================
    v16h aI[2], aP[2];
#pragma unroll
    for (int k = 0; k < 2; ++k) {
      aI[k] = load_a_lds(sH0 + mt * 16 * kH + k * 32, kH);
      aP[k] = load_a_lds(sH1 + mt * 16 * kH + k * 32, kH);
    }
    float hn1[8];
    {
      v8f ar, az, ain, ahn;
#pragma unroll
      for (int v = 0; v < 8; ++v) { ar[v] = bR1; az[v] = bZ1; ain[v] = bNi1; ahn[v] = bNh1; }
#pragma unroll
      for (int k = 0; k < 2; ++k) {
        ar  = wmma16(aI[k], load_b_nmajor(sWih1 + (nt * 16) * kH + k * 32, kH), ar);
        az  = wmma16(aI[k], load_b_nmajor(sWih1 + (kH + nt * 16) * kH + k * 32, kH), az);
        ain = wmma16(aI[k], load_b_nmajor(sWih1 + (2 * kH + nt * 16) * kH + k * 32, kH), ain);
        // W_hh1 B-fragments streamed from global (contiguous 64B/lane, L2-resident)
        ar  = wmma16(aP[k], load_b_nmajor_g(Whh1 + (nt * 16) * kH + k * 32, kH), ar);
        az  = wmma16(aP[k], load_b_nmajor_g(Whh1 + (kH + nt * 16) * kH + k * 32, kH), az);
        ahn = wmma16(aP[k], load_b_nmajor_g(Whh1 + (2 * kH + nt * 16) * kH + k * 32, kH), ahn);
      }
#pragma unroll
      for (int v = 0; v < 8; ++v) {
        float r_ = sigm_(ar[v]);
        float z_ = sigm_(az[v]);
        float n_ = tanhf(ain[v] + r_ * ahn[v]);
        float hp = (float)sH1[(rowBase + v) * kH + nt * 16 + colL];
        hn1[v] = (1.0f - z_) * n_ + z_ * hp;
      }
    }
    __syncthreads();
#pragma unroll
    for (int v = 0; v < 8; ++v) {
      sH1[(rowBase + v) * kH + nt * 16 + colL] = (_Float16)hn1[v];
      if (t == kT - 1)
        hidden[(size_t)(n0 + rowBase + v) * kH + nt * 16 + colL] = hn1[v];
    }
    __syncthreads();
  }

  // ================= hp = hidden @ Wt^T + bt ; s_src / s_dst =================
  _Float16* WtS = reinterpret_cast<_Float16*>(sXf);   // reuse x buffer (8 KB of 20 KB)
  float*    sHP = reinterpret_cast<float*>(sWih0);    // reuse Wih0 buffer (8 KB of 60 KB)
  for (int idx = tid; idx < kH * kH; idx += 256) WtS[idx] = (_Float16)Wt[idx];
  __syncthreads();

  v16h aF[2];
#pragma unroll
  for (int k = 0; k < 2; ++k) aF[k] = load_a_lds(sH1 + mt * 16 * kH + k * 32, kH);
  {
    v8f acc;
    float bc = bt[cB];
#pragma unroll
    for (int v = 0; v < 8; ++v) acc[v] = bc;
#pragma unroll
    for (int k = 0; k < 2; ++k)
      acc = wmma16(aF[k], load_b_nmajor(WtS + (nt * 16) * kH + k * 32, kH), acc);
#pragma unroll
    for (int v = 0; v < 8; ++v) sHP[(rowBase + v) * kH + cB] = acc[v];
  }
  __syncthreads();

  if (tid < kRows) {
    float ss = 0.0f, sd = 0.0f;
    for (int j = 0; j < kH; ++j) {
      float hv = sHP[tid * kH + j];
      ss += hv * avec[j];
      sd += hv * avec[kH + j];
    }
    s_src[n0 + tid] = ss;
    s_dst[n0 + tid] = sd;
  }
}

// ======================= Kernel 2: flash-softmax attention + FC head =======================
__global__ __launch_bounds__(256) void attn_head_kernel(
    const float* __restrict__ hidden, const float* __restrict__ s_src,
    const float* __restrict__ s_dst,
    const float* __restrict__ Wfc, const float* __restrict__ bfc,
    const float* __restrict__ Wout, const float* __restrict__ bout,
    float* __restrict__ out)
{
  __shared__ __attribute__((aligned(16))) float sHbuf[2][32 * kH]; // 16 KB async hidden chunks
  __shared__ _Float16 sA[64 * kH];    // h2 (post-attention) f16
  __shared__ _Float16 sB[64 * kH];    // h3 (post-fc) f16
  __shared__ _Float16 sW[64 * kH];    // weight staging
  __shared__ float sRed[256];
  __shared__ float sSrcC[32];
  __shared__ float sDstL[64];
  __shared__ float sZ[64];
  __shared__ float sMax;

  const int tid  = threadIdx.x;
  const int wave = tid >> 5;
  const int lane = tid & 31;
  const int i0   = blockIdx.x * 64;

  const int mt      = wave >> 1;
  const int colL    = lane & 15;
  const int rowBase = mt * 16 + ((lane >> 4) << 3);

  const unsigned hlds0 = lds_off_u32(&sHbuf[0][0]);
  const unsigned hlds1 = lds_off_u32(&sHbuf[1][0]);

  // kick off async stage of hidden chunk 0 (8 KB contiguous)
  for (int k = 0; k < 8; ++k) {
    unsigned idx = (unsigned)(tid + k * 256);
    async_g2l_b32(hlds0 + idx * 4u, idx * 4u, hidden);
  }

  // global max of s_src (leaky monotone => row max = leaky(Smax + s_dst[i]))
  float m = -1e30f;
  for (int j = tid; j < kN; j += 256) m = fmaxf(m, s_src[j]);
  sRed[tid] = m;
  __syncthreads();
  for (int off = 128; off > 0; off >>= 1) {
    if (tid < off) sRed[tid] = fmaxf(sRed[tid], sRed[tid + off]);
    __syncthreads();
  }
  if (tid == 0) sMax = sRed[0];
  if (tid < 64) sDstL[tid] = s_dst[i0 + tid];
  __syncthreads();

  const float myDst = sDstL[mt * 16 + (lane & 15)];
  const float m_i   = leaky_(sMax + myDst);

  v8f acc[2];
#pragma unroll
  for (int s = 0; s < 2; ++s)
#pragma unroll
    for (int v = 0; v < 8; ++v) acc[s][v] = 0.0f;
  float zpart = 0.0f;

  for (int c = 0; c < kN / 32; ++c) {
    wait_async0();
    __syncthreads();                       // chunk c landed; prev-iter LDS reads done
    if (tid < 32) sSrcC[tid] = s_src[c * 32 + tid];
    if (c + 1 < kN / 32) {                 // overlap: fetch chunk c+1 during compute
      unsigned dst = ((c + 1) & 1) ? hlds1 : hlds0;
      unsigned gbase = (unsigned)((c + 1) * 32 * kH);
      for (int k = 0; k < 8; ++k) {
        unsigned idx = (unsigned)(tid + k * 256);
        async_g2l_b32(dst + idx * 4u, (gbase + idx) * 4u, hidden);
      }
    }
    __syncthreads();                       // sSrcC visible

    const float* hc = &sHbuf[c & 1][0];
    const int kb = (lane >> 4) << 4;
    v16h aW;
#pragma unroll
    for (int i = 0; i < 8; ++i) {
      float w0 = __expf(leaky_(sSrcC[kb + i] + myDst) - m_i);
      float w1 = __expf(leaky_(sSrcC[kb + 8 + i] + myDst) - m_i);
      aW[i]     = (_Float16)w0;
      aW[i + 8] = (_Float16)w1;
      zpart += w0 + w1;
    }
#pragma unroll
    for (int s = 0; s < 2; ++s) {
      const int nt = ((wave & 1) << 1) + s;
      acc[s] = wmma16(aW, load_b_kmajor_f32(hc + nt * 16, kH), acc[s]);
    }
  }

  float zfull = zpart + __shfl_xor(zpart, 16, 32);
  if ((wave & 1) == 0 && lane < 16) sZ[mt * 16 + lane] = zfull;
  __syncthreads();

  // h2 = att@hidden / Z + hidden (residual)
#pragma unroll
  for (int s = 0; s < 2; ++s) {
    const int nt = ((wave & 1) << 1) + s;
#pragma unroll
    for (int v = 0; v < 8; ++v) {
      int rw = rowBase + v, cl = nt * 16 + colL;
      float hv = hidden[(size_t)(i0 + rw) * kH + cl];
      sA[rw * kH + cl] = (_Float16)(acc[s][v] / sZ[rw] + hv);
    }
  }
  for (int idx = tid; idx < kH * kH; idx += 256) sW[idx] = (_Float16)Wfc[idx];
  __syncthreads();

  // h3 = leaky(h2 @ Wfc^T + bfc)
  v16h aF[2];
#pragma unroll
  for (int k = 0; k < 2; ++k) aF[k] = load_a_lds(sA + mt * 16 * kH + k * 32, kH);
#pragma unroll
  for (int s = 0; s < 2; ++s) {
    const int nt = ((wave & 1) << 1) + s;
    const int cc = nt * 16 + colL;
    v8f a2;
    float bc = bfc[cc];
#pragma unroll
    for (int v = 0; v < 8; ++v) a2[v] = bc;
#pragma unroll
    for (int k = 0; k < 2; ++k)
      a2 = wmma16(aF[k], load_b_nmajor(sW + (nt * 16) * kH + k * 32, kH), a2);
#pragma unroll
    for (int v = 0; v < 8; ++v)
      sB[(rowBase + v) * kH + cc] = (_Float16)leaky_(a2[v]);
  }
  __syncthreads();
  for (int idx = tid; idx < kH * kH; idx += 256) sW[idx] = (_Float16)Wout[idx];
  __syncthreads();

  // out = h3 @ Wout^T + bout
  v16h aO[2];
#pragma unroll
  for (int k = 0; k < 2; ++k) aO[k] = load_a_lds(sB + mt * 16 * kH + k * 32, kH);
#pragma unroll
  for (int s = 0; s < 2; ++s) {
    const int nt = ((wave & 1) << 1) + s;
    const int cc = nt * 16 + colL;
    v8f a3;
    float bc = bout[cc];
#pragma unroll
    for (int v = 0; v < 8; ++v) a3[v] = bc;
#pragma unroll
    for (int k = 0; k < 2; ++k)
      a3 = wmma16(aO[k], load_b_nmajor(sW + (nt * 16) * kH + k * 32, kH), a3);
#pragma unroll
    for (int v = 0; v < 8; ++v)
      out[(size_t)(i0 + rowBase + v) * kH + cc] = a3[v];
  }
}

extern "C" void kernel_launch(void* const* d_in, const int* in_sizes, int n_in,
                              void* d_out, int out_size, void* d_ws, size_t ws_size,
                              hipStream_t stream) {
  const float* x    = (const float*)d_in[0];
  const float* Wih0 = (const float*)d_in[1];
  const float* Whh0 = (const float*)d_in[2];
  const float* bih0 = (const float*)d_in[3];
  const float* bhh0 = (const float*)d_in[4];
  const float* Wih1 = (const float*)d_in[5];
  const float* Whh1 = (const float*)d_in[6];
  const float* bih1 = (const float*)d_in[7];
  const float* bhh1 = (const float*)d_in[8];
  const float* Wt   = (const float*)d_in[9];
  const float* bt   = (const float*)d_in[10];
  const float* avec = (const float*)d_in[11];
  const float* Wfc  = (const float*)d_in[12];
  const float* bfc  = (const float*)d_in[13];
  const float* Wout = (const float*)d_in[14];
  const float* bout = (const float*)d_in[15];

  float* hidden = (float*)d_ws;               // N*H floats
  float* ssrc   = hidden + (size_t)kN * kH;   // N floats
  float* sdst   = ssrc + kN;                  // N floats
  float* outp   = (float*)d_out;              // N*H floats

  gru_fused_kernel<<<kN / kRows, 256, 0, stream>>>(
      x, Wih0, Whh0, bih0, bhh0, Wih1, Whh1, bih1, bhh1, Wt, bt, avec,
      hidden, ssrc, sdst);
  attn_head_kernel<<<kN / 64, 256, 0, stream>>>(
      hidden, ssrc, sdst, Wfc, bfc, Wout, bout, outp);
}
